// WordBiLSTMCRF_9844065042754
// MI455X (gfx1250) — matrix operand order repaired and checked by reading, not compiled, not run on hardware
//
#include <hip/hip_runtime.h>
#include <hip/hip_bf16.h>

// ---------------------------------------------------------------- types
typedef __attribute__((ext_vector_type(16))) __bf16 v16bf;
typedef __attribute__((ext_vector_type(8)))  float  v8f;

#define TT 2048
#define EE 512
#define GG 2048   // 4*HD gate rows
#define HD 512
#define NWG 16    // workgroups per direction in the recurrence
#define HPW 32    // h elements owned per workgroup
#define ROWS_PER_WG 128   // 4 gates x HPW
#define WROWU32 260       // padded LDS row stride in dwords (1024B + 16B TDM pad)
#define NTAGS 32
#define TAG_START 30
#define TAG_STOP 31

__device__ __forceinline__ unsigned short f2bf(float f) {
    unsigned u = __float_as_uint(f);
    unsigned r = u + 0x7FFFu + ((u >> 16) & 1u);   // round-nearest-even
    return (unsigned short)(r >> 16);
}
__device__ __forceinline__ float bflo(unsigned u) { return __uint_as_float(u << 16); }
__device__ __forceinline__ float bfhi(unsigned u) { return __uint_as_float(u & 0xffff0000u); }
__device__ __forceinline__ float sigf(float x) { return 1.0f / (1.0f + __expf(-x)); }

// ---------------------------------------------------------------- prep
__global__ __launch_bounds__(256) void conv_bf16(const float* __restrict__ s,
                                                 unsigned short* __restrict__ d, int n) {
    int i = blockIdx.x * blockDim.x + threadIdx.x;
    if (i < n) d[i] = f2bf(s[i]);
}

__global__ __launch_bounds__(256) void gather_bf16(const int* __restrict__ sent,
                                                   const float* __restrict__ emb,
                                                   unsigned short* __restrict__ xbf) {
    int i = blockIdx.x * blockDim.x + threadIdx.x;   // T*E threads
    int t = i >> 9, e = i & 511;
    if (t < TT) xbf[i] = f2bf(emb[(size_t)sent[t] * EE + e]);
}

__global__ __launch_bounds__(1024) void init_state(const float* __restrict__ h0,
                                                   unsigned short* __restrict__ hexch,
                                                   unsigned int* __restrict__ cnt) {
    int tid = threadIdx.x;
    if (tid < 1024) {
        int dir = tid >> 9, j = tid & 511;
        hexch[(dir * 2 + 1) * HD + j] = f2bf(h0[tid]);  // "previous h" buffer for step 0
    }
    if (tid < 2) cnt[tid * 64] = 0u;                    // per-direction sync counters
}

// ---------------------------------------------------------------- WMMA input GEMM
// xg[dir][t][g] = sum_k x[t][k] * w_ih[dir][g][k] + b[dir][g]
__global__ __launch_bounds__(256) void xgemm_wmma(const unsigned short* __restrict__ xbf,
                                                  const unsigned short* __restrict__ wihbf,
                                                  const float* __restrict__ bias_f,
                                                  const float* __restrict__ bias_b,
                                                  float* __restrict__ xg) {
    int wave = (blockIdx.x * blockDim.x + threadIdx.x) >> 5;
    int lane = threadIdx.x & 31;
    int nt  = wave & 127;
    int mt  = (wave >> 7) & 127;
    int dir = wave >> 14;
    const unsigned short* wdir = wihbf + (size_t)dir * GG * EE;
    const float* bias = dir ? bias_b : bias_f;

    int m0 = mt * 16, n0 = nt * 16;
    int mrow = m0 + (lane & 15);
    int nrow = n0 + (lane & 15);
    int kbA = (lane >> 4) * 8;    // A: two 8-elt chunks at kbA and kbA+16
    int kbB = (lane >> 4) * 8;    // B: two 8-elt chunks at kbB and kbB+16

    v8f acc = {};
    for (int k0 = 0; k0 < EE; k0 += 32) {
        union { v16bf v; uint4 q[2]; } a, b;
        const unsigned short* ap = xbf + (size_t)mrow * EE + k0 + kbA;
        a.q[0] = *(const uint4*)(ap);
        a.q[1] = *(const uint4*)(ap + 16);
        const unsigned short* bp = wdir + (size_t)nrow * EE + k0 + kbB;
        b.q[0] = *(const uint4*)(bp);
        b.q[1] = *(const uint4*)(bp + 16);
        acc = __builtin_amdgcn_wmma_f32_16x16x32_bf16(false, a.v, false, b.v,
                                                      (short)0, acc, false, false);
    }
    float bv = bias[nrow];
    int mhalf = (lane >> 4) * 8;
    float* outp = xg + ((size_t)dir * TT + m0) * GG + n0 + (lane & 15);
#pragma unroll
    for (int r = 0; r < 8; ++r)
        outp[(size_t)(mhalf + r) * GG] = acc[r] + bv;
}

// ---------------------------------------------------------------- recurrence
// 2 dirs x NWG blocks; block owns 32 h elems (128 gate rows) with its 128KB w_hh
// slab TDM-loaded into LDS once (pad 16B per 1KB row to skew banks). 2 threads
// split each row's K=512 dot; h (512 bf16) is broadcast-read from LDS.
__global__ __launch_bounds__(256) void lstm_rec(const unsigned short* __restrict__ whhbf,
                                                const float* __restrict__ xg,
                                                const float* __restrict__ c0,
                                                unsigned short* __restrict__ hexch,
                                                unsigned short* __restrict__ hsbuf,
                                                unsigned int* __restrict__ cnt) {
    extern __shared__ unsigned int wlds[];   // ROWS_PER_WG x WROWU32 dwords
    __shared__ unsigned int hsh[HD / 2];     // 512 bf16 packed as 256 dwords
    __shared__ float gsh[256];
    __shared__ float csh[HPW];

    int dir = blockIdx.x / NWG;
    int wg  = blockIdx.x % NWG;
    int tid = threadIdx.x;
    int jb  = wg * HPW;

    const unsigned short* gsrc = whhbf + ((size_t)dir * GG + jb) * HD;  // i-gate row jb

#if __has_builtin(__builtin_amdgcn_tensor_load_to_lds)
    if (tid < 32) {   // one wave issues the TDM descriptor for the whole slab
        typedef __attribute__((ext_vector_type(4))) unsigned int u32x4;
        typedef __attribute__((ext_vector_type(8))) int i32x8;
        typedef __attribute__((ext_vector_type(4))) int i32x4;
        unsigned long long ga = (unsigned long long)(size_t)gsrc;
        unsigned ldsb = (unsigned)(size_t)(void*)wlds;
        u32x4 g0 = {0u, 0u, 0u, 0u};
        g0[0] = 1u;                                       // count=1 (valid user D#)
        g0[1] = ldsb;                                     // lds_addr (bytes)
        g0[2] = (unsigned)(ga & 0xffffffffu);             // global_addr[31:0]
        g0[3] = (unsigned)((ga >> 32) & 0x01ffffffu)      // global_addr[56:32]
              | (2u << 30);                               // type=2 ("image")
        i32x8 g1 = {0, 0, 0, 0, 0, 0, 0, 0};
        const unsigned td0 = 32u * HD;                    // 16384-elem gate slice line
        const unsigned long long st0 = (unsigned long long)HD * HD;  // 262144 elems
        g1[0] = (int)((1u << 16)        // data_size = 2 bytes
                    | (1u << 20)        // pad_enable
                    | (7u << 22)        // pad_interval: every 256 dwords (1KB)
                    | (3u << 25));      // pad_amount: 4 dwords (16B)
        g1[1] = (int)((td0 & 0xffffu) << 16);             // tensor_dim0[15:0]
        g1[2] = (int)((td0 >> 16) | (4u << 16));          // tensor_dim0[31:16]|tensor_dim1=4
        g1[3] = (int)(td0 << 16);                         // tile_dim0 = 16384
        g1[4] = (int)4;                                   // tile_dim1=4, tile_dim2=0
        g1[5] = (int)(st0 & 0xffffffffu);                 // tensor_dim0_stride lo
        g1[6] = (int)((st0 >> 32) & 0xffffu);             // stride hi | dim1_stride lo=0
        g1[7] = 0;
        i32x4 g2 = {0, 0, 0, 0}, g3 = {0, 0, 0, 0};
#if defined(__clang_major__) && __clang_major__ >= 23
        i32x8 g4 = {0, 0, 0, 0, 0, 0, 0, 0};
        __builtin_amdgcn_tensor_load_to_lds(g0, g1, g2, g3, g4, 0);
#else
        __builtin_amdgcn_tensor_load_to_lds(g0, g1, g2, g3, 0);
#endif
        __builtin_amdgcn_s_wait_tensorcnt(0);
    }
#else
    // cooperative fallback copy replicating the padded layout
    for (int i = tid; i < ROWS_PER_WG * WROWU32; i += 256) {
        int r = i / WROWU32, w = i % WROWU32;
        int gate = r >> 5, jl = r & 31;
        const unsigned* src =
            (const unsigned*)(whhbf + ((size_t)dir * GG + gate * HD + jb + jl) * HD);
        wlds[i] = (w < 256) ? src[w] : 0u;
    }
#endif
    if (tid < HPW) csh[tid] = c0[dir * HD + jb + tid];
    __syncthreads();

    int half = tid >> 7;              // which K half (0: k<256, 1: k>=256)
    int r    = tid & 127;             // local gate row; gate = r>>5, jl = r&31
    int grow = (r >> 5) * HD + jb + (r & 31);
    const float* xgd = xg + (size_t)dir * TT * GG;
    unsigned short* hex = hexch + dir * 2 * HD;
    unsigned int* mycnt = cnt + dir * 64;
    const uint4* wrow = (const uint4*)wlds + (size_t)r * (WROWU32 / 4) + half * 64;
    const unsigned int* hbase = hsh + half * 128;

    for (int s = 0; s < TT; ++s) {
        int t = dir ? (TT - 1 - s) : s;
        // pull full previous h (all blocks' slices) into LDS
        hsh[tid] = ((const unsigned int*)(hex + (((s & 1) ^ 1) * HD)))[tid];
        __syncthreads();

        float acc = (half == 0) ? xgd[(size_t)t * GG + grow] : 0.0f;
        if (half == 0 && s + 1 < TT)
            __builtin_prefetch(xgd + (size_t)(dir ? t - 1 : t + 1) * GG + grow, 0, 1);
#pragma unroll 4
        for (int c = 0; c < 32; ++c) {    // 32 x 8 = 256 MACs (this thread's K half)
            uint4 w = wrow[c];
            unsigned ha = hbase[c * 4 + 0], hb = hbase[c * 4 + 1];
            unsigned hc = hbase[c * 4 + 2], hd = hbase[c * 4 + 3];
            acc = fmaf(bflo(w.x), bflo(ha), acc);
            acc = fmaf(bfhi(w.x), bfhi(ha), acc);
            acc = fmaf(bflo(w.y), bflo(hb), acc);
            acc = fmaf(bfhi(w.y), bfhi(hb), acc);
            acc = fmaf(bflo(w.z), bflo(hc), acc);
            acc = fmaf(bfhi(w.z), bfhi(hc), acc);
            acc = fmaf(bflo(w.w), bflo(hd), acc);
            acc = fmaf(bfhi(w.w), bfhi(hd), acc);
        }
        gsh[tid] = acc;
        __syncthreads();

        if (tid < HPW) {
            float ig = gsh[tid]      + gsh[128 + tid];
            float fg = gsh[32 + tid] + gsh[160 + tid];
            float gg = gsh[64 + tid] + gsh[192 + tid];
            float og = gsh[96 + tid] + gsh[224 + tid];
            float c  = sigf(fg) * csh[tid] + sigf(ig) * tanhf(gg);
            float h  = sigf(og) * tanhf(c);
            csh[tid] = c;
            unsigned short hb16 = f2bf(h);
            hex[(s & 1) * HD + jb + tid] = hb16;                    // exchange slice
            hsbuf[(size_t)t * 1024 + dir * HD + jb + tid] = hb16;   // concat(hf,hb) row
        }
        __syncthreads();
        if (tid == 0) {
            __threadfence();
            atomicAdd(mycnt, 1u);
            unsigned target = (unsigned)(s + 1) * NWG;
            while (__hip_atomic_load(mycnt, __ATOMIC_ACQUIRE, __HIP_MEMORY_SCOPE_AGENT) < target)
                __builtin_amdgcn_s_sleep(1);
        }
        __syncthreads();
    }
}

// ---------------------------------------------------------------- WMMA feature GEMM
// feats[t][tag] = sum_k hs[t][k] * W_out[tag][k] + b_out[tag]
__global__ __launch_bounds__(256) void fgemm_wmma(const unsigned short* __restrict__ hs,
                                                  const unsigned short* __restrict__ wobf,
                                                  const float* __restrict__ bout,
                                                  float* __restrict__ feats) {
    int wave = (blockIdx.x * blockDim.x + threadIdx.x) >> 5;
    int lane = threadIdx.x & 31;
    int nt = wave & 1;      // 32 tags -> 2 N tiles
    int mt = wave >> 1;     // 128 M tiles
    int m0 = mt * 16, n0 = nt * 16;
    int mrow = m0 + (lane & 15);
    int nrow = n0 + (lane & 15);
    int kbA = (lane >> 4) * 8;
    int kbB = (lane >> 4) * 8;

    v8f acc = {};
    for (int k0 = 0; k0 < 1024; k0 += 32) {
        union { v16bf v; uint4 q[2]; } a, b;
        const unsigned short* ap = hs + (size_t)mrow * 1024 + k0 + kbA;
        a.q[0] = *(const uint4*)(ap);
        a.q[1] = *(const uint4*)(ap + 16);
        const unsigned short* bp = wobf + (size_t)nrow * 1024 + k0 + kbB;
        b.q[0] = *(const uint4*)(bp);
        b.q[1] = *(const uint4*)(bp + 16);
        acc = __builtin_amdgcn_wmma_f32_16x16x32_bf16(false, a.v, false, b.v,
                                                      (short)0, acc, false, false);
    }
    float bv = bout[nrow];
    int mhalf = (lane >> 4) * 8;
    float* outp = feats + (size_t)(m0)*NTAGS + n0 + (lane & 15);
#pragma unroll
    for (int r = 0; r < 8; ++r)
        outp[(size_t)(mhalf + r) * NTAGS] = acc[r] + bv;
}

// ---------------------------------------------------------------- Viterbi (1 wave)
__global__ __launch_bounds__(32) void viterbi_k(const float* __restrict__ feats,
                                                const float* __restrict__ trans,
                                                unsigned char* __restrict__ bp,
                                                float* __restrict__ out) {
    int n = threadIdx.x;  // tag id, 32 lanes
    __shared__ float fvs[NTAGS];
    __shared__ float term[NTAGS];
    float tr[NTAGS];
#pragma unroll
    for (int p = 0; p < NTAGS; ++p) tr[p] = trans[n * NTAGS + p];
    float fv = (n == TAG_START) ? 0.0f : -10000.0f;
    for (int t = 0; t < TT; ++t) {
        fvs[n] = fv;
        __syncthreads();
        float best = fvs[0] + tr[0];
        int bi = 0;
#pragma unroll
        for (int p = 1; p < NTAGS; ++p) {
            float sc = fvs[p] + tr[p];
            if (sc > best) { best = sc; bi = p; }   // first-max tie-break like argmax
        }
        bp[t * NTAGS + n] = (unsigned char)bi;
        fv = best + feats[t * NTAGS + n];
        __syncthreads();
    }
    term[n] = fv + trans[TAG_STOP * NTAGS + n];
    __syncthreads();
    if (n == 0) {
        int bt = 0;
        float bs = term[0];
        for (int p = 1; p < NTAGS; ++p)
            if (term[p] > bs) { bs = term[p]; bt = p; }
        out[0] = bs;
        int tag = bt;
        for (int t = TT - 1; t >= 0; --t) {
            out[1 + t] = (float)tag;
            tag = bp[t * NTAGS + tag];
        }
    }
}

// ---------------------------------------------------------------- launch
extern "C" void kernel_launch(void* const* d_in, const int* in_sizes, int n_in,
                              void* d_out, int out_size, void* d_ws, size_t ws_size,
                              hipStream_t stream) {
    const int*   sentence = (const int*)d_in[0];
    const float* emb      = (const float*)d_in[1];
    const float* w_ih_f   = (const float*)d_in[2];
    const float* w_hh_f   = (const float*)d_in[3];
    const float* b_f      = (const float*)d_in[4];
    const float* w_ih_b   = (const float*)d_in[5];
    const float* w_hh_b   = (const float*)d_in[6];
    const float* b_b      = (const float*)d_in[7];
    const float* W_out    = (const float*)d_in[8];
    const float* b_out    = (const float*)d_in[9];
    const float* trans    = (const float*)d_in[10];
    const float* h0       = (const float*)d_in[11];
    const float* c0       = (const float*)d_in[12];
    (void)in_sizes; (void)n_in; (void)out_size; (void)ws_size;

    // carve workspace (256B aligned)
    size_t off = 0;
    char* w = (char*)d_ws;
    auto carve = [&](size_t b) { size_t o = off; off = (off + b + 255) & ~(size_t)255; return o; };
    unsigned short* x_bf   = (unsigned short*)(w + carve((size_t)TT * EE * 2));
    unsigned short* wih_bf = (unsigned short*)(w + carve((size_t)2 * GG * EE * 2));
    unsigned short* whh_bf = (unsigned short*)(w + carve((size_t)2 * GG * EE * 2));
    unsigned short* wo_bf  = (unsigned short*)(w + carve((size_t)NTAGS * 1024 * 2));
    float*          xg     = (float*)(w + carve((size_t)2 * TT * GG * 4));
    unsigned short* hsbuf  = (unsigned short*)(w + carve((size_t)TT * 1024 * 2));
    unsigned short* hexch  = (unsigned short*)(w + carve((size_t)2 * 2 * HD * 2));
    float*          feats  = (float*)(w + carve((size_t)TT * NTAGS * 4));
    unsigned char*  bp     = (unsigned char*)(w + carve((size_t)TT * NTAGS));
    unsigned int*   cnt    = (unsigned int*)(w + carve(512));

    const int WE = GG * EE;  // 1M elements per weight matrix
    conv_bf16<<<(WE + 255) / 256, 256, 0, stream>>>(w_ih_f, wih_bf, WE);
    conv_bf16<<<(WE + 255) / 256, 256, 0, stream>>>(w_ih_b, wih_bf + (size_t)WE, WE);
    conv_bf16<<<(WE + 255) / 256, 256, 0, stream>>>(w_hh_f, whh_bf, WE);
    conv_bf16<<<(WE + 255) / 256, 256, 0, stream>>>(w_hh_b, whh_bf + (size_t)WE, WE);
    conv_bf16<<<(NTAGS * 1024 + 255) / 256, 256, 0, stream>>>(W_out, wo_bf, NTAGS * 1024);

    gather_bf16<<<(TT * EE) / 256, 256, 0, stream>>>(sentence, emb, x_bf);
    init_state<<<1, 1024, 0, stream>>>(h0, hexch, cnt);

    // 2 dirs x 128 x 128 tiles = 32768 waves, 8 waves/block
    xgemm_wmma<<<4096, 256, 0, stream>>>(x_bf, wih_bf, b_f, b_b, xg);

    // persistent recurrence: 32 blocks, 130KB dynamic LDS each (w_hh slab, padded)
    lstm_rec<<<2 * NWG, 256, ROWS_PER_WG * WROWU32 * 4, stream>>>(whh_bf, xg, c0,
                                                                  hexch, hsbuf, cnt);

    // 128 x 2 tiles = 256 waves, 8 waves/block
    fgemm_wmma<<<32, 256, 0, stream>>>(hsbuf, wo_bf, b_out, feats);

    viterbi_k<<<1, 32, 0, stream>>>(feats, trans, bp, (float*)d_out);
}
